// DKT_67585605369960
// MI455X (gfx1250) — matrix operand order repaired and checked by reading, not compiled, not run on hardware
//
#include <hip/hip_runtime.h>
#include <math.h>

typedef __attribute__((ext_vector_type(16))) __bf16 v16bf;
typedef __attribute__((ext_vector_type(8)))  float  v8f;

union Frag16 {
    v16bf v;
    unsigned short u[16];
    uint4 q[2];
};

__device__ __forceinline__ unsigned short f2b(float f) {
    unsigned int x = __float_as_uint(f);
    unsigned int r = x + 0x7FFFu + ((x >> 16) & 1u);   // RNE
    return (unsigned short)(r >> 16);
}
__device__ __forceinline__ float sigf(float x) {
    return 1.0f / (1.0f + expf(-x));
}

// ---------------- small utility kernels ----------------

// Pack an f32 weight matrix (optionally transposed on read) into the WMMA
// B-fragment bf16 layout in this code's contiguous k order: dst[((tk*ntiles + tn)*32 + lane)*16 + e]
//   = B[tk*32 + (lane>>4)*16 + e][tn*16 + (lane&15)]
// so a wave's 32x16 B fragment is 32 contiguous bytes per lane (2x b128 loads).
__global__ void pack_b_kernel(const float* __restrict__ src, unsigned short* __restrict__ dst,
                              int K, int Nc, int srcTransposed) {
    int i = blockIdx.x * blockDim.x + threadIdx.x;
    if (i >= K * Nc) return;
    int e    = i & 15;
    int lane = (i >> 4) & 31;
    int rest = i >> 9;
    int ntiles = Nc >> 4;
    int tn = rest % ntiles;
    int tk = rest / ntiles;
    int krow = (tk << 5) + ((lane >> 4) << 4) + e;
    int col  = (tn << 4) + (lane & 15);
    float v = srcTransposed ? src[(size_t)col * K + krow]
                            : src[(size_t)krow * Nc + col];
    dst[i] = f2b(v);
}

__global__ void add_bias_kernel(const float* __restrict__ a, const float* __restrict__ b,
                                float* __restrict__ out, int n) {
    int i = blockIdx.x * blockDim.x + threadIdx.x;
    if (i < n) out[i] = a[i] + b[i];
}

// Deterministic CSR: one thread per destination node scans the full edge list in order.
__global__ void build_csr_kernel(const int* __restrict__ src, const int* __restrict__ dst,
                                 int E, int Nn, int* __restrict__ deg, int* __restrict__ adj) {
    int n = blockIdx.x * blockDim.x + threadIdx.x;
    if (n >= Nn) return;
    int d = 0;
    for (int e = 0; e < E; ++e) {
        if (dst[e] == n && d < 8) { adj[n * 8 + d] = src[e]; ++d; }
    }
    deg[n] = d;
}

// Xin[bn, 0:256] = emb_p[p[bn]], [256:512] = emb_aff[aff[bn]]  (bf16)
__global__ void build_xin_kernel(const int* __restrict__ p, const int* __restrict__ aff,
                                 const float* __restrict__ emb_p, const float* __restrict__ emb_aff,
                                 unsigned short* __restrict__ Xin, int BN) {
    int i = blockIdx.x * blockDim.x + threadIdx.x;
    if (i >= BN * 512) return;
    int bn = i >> 9;
    int c = i & 511;
    float v = (c < 256) ? emb_p[(size_t)p[bn] * 256 + c]
                        : emb_aff[(size_t)aff[bn] * 256 + (c - 256)];
    Xin[i] = f2b(v);
}

// Seq[bn, 0:768] = [p_emb | q_emb | r_emb] (bf16); slot [768:1024] filled by GAT2 aggregate.
__global__ void build_seq_kernel(const int* __restrict__ p, const int* __restrict__ q,
                                 const int* __restrict__ r,
                                 const float* __restrict__ emb_p, const float* __restrict__ emb_q,
                                 const float* __restrict__ emb_r,
                                 unsigned short* __restrict__ Seq, int BN) {
    int i = blockIdx.x * blockDim.x + threadIdx.x;
    if (i >= BN * 768) return;
    int bn = i / 768;
    int c = i - bn * 768;
    float v;
    if (c < 256)      v = emb_p[(size_t)p[bn] * 256 + c];
    else if (c < 512) v = emb_q[(size_t)q[bn] * 256 + (c - 256)];
    else              v = emb_r[(size_t)r[bn] * 256 + (c - 512)];
    Seq[(size_t)bn * 1024 + c] = f2b(v);
}

// ---------------- WMMA bf16 GEMM: C[M,Nc] = A[M,K] @ Bpacked (+bias) ----------------
// Block: 256 threads = 8 waves. Block tile 64M x 128N; A tile (64x32) staged in LDS
// once per K-step and shared by all 8 waves; B fragments are contiguous (packed).
#define ALDS_STRIDE 40   // 32 payload + 8 pad ushorts: keeps b128 alignment, spreads banks
__global__ __launch_bounds__(256)
void gemm_bf16_kernel(const unsigned short* __restrict__ A,
                      const unsigned short* __restrict__ Bp,
                      const float* __restrict__ bias,
                      float* __restrict__ Cf,
                      unsigned short* __restrict__ Cb,
                      int M, int Nc, int K) {
    __shared__ __align__(16) unsigned short Alds[64 * ALDS_STRIDE];

    const int tid  = threadIdx.x;
    const int lane = tid & 31;
    const int wave = tid >> 5;
    const int m0 = blockIdx.x * 64;
    const int n0 = blockIdx.y * 128 + wave * 16;
    const int l15 = lane & 15;
    const int lhi = lane >> 4;
    const int col = n0 + l15;
    const int ntiles = Nc >> 4;
    const int tn = n0 >> 4;

    const int arow = tid >> 2;          // 0..63
    const int achk = (tid & 3) << 3;    // 0,8,16,24
    const unsigned short* Aouter = A + (size_t)(m0 + arow) * K + achk;

    v8f acc[4];
    acc[0] = {}; acc[1] = {}; acc[2] = {}; acc[3] = {};

    for (int k0 = 0; k0 < K; k0 += 32) {
        __builtin_prefetch(Aouter + k0 + 32, 0, 1);  // global_prefetch of next A tile
        __syncthreads();                              // previous iteration's reads done
        *(uint4*)&Alds[arow * ALDS_STRIDE + achk] = *(const uint4*)(Aouter + k0);
        __syncthreads();

        Frag16 bf;
        {
            const unsigned short* bp =
                Bp + (((((size_t)(k0 >> 5)) * ntiles + tn) * 32 + lane) << 4);
            bf.q[0] = *(const uint4*)(bp);
            bf.q[1] = *(const uint4*)(bp + 8);
        }
#pragma unroll
        for (int mi = 0; mi < 4; ++mi) {
            Frag16 af;
            const unsigned short* ap = &Alds[(mi * 16 + l15) * ALDS_STRIDE + (lhi << 4)];
            af.q[0] = *(const uint4*)(ap);
            af.q[1] = *(const uint4*)(ap + 8);
            acc[mi] = __builtin_amdgcn_wmma_f32_16x16x32_bf16(
                false, af.v, false, bf.v, (short)0, acc[mi], false, false);
        }
    }

    const float bv = bias ? bias[col] : 0.0f;
#pragma unroll
    for (int mi = 0; mi < 4; ++mi) {
        const int rbase = m0 + mi * 16 + (lhi << 3);
#pragma unroll
        for (int v = 0; v < 8; ++v) {
            const int row = rbase + v;
            if (row < M) {
                const float val = acc[mi][v] + bv;
                const size_t idx = (size_t)row * Nc + col;
                if (Cf) Cf[idx] = val;
                if (Cb) Cb[idx] = f2b(val);
            }
        }
    }
}

// ---------------- GAT attention scores: es/ed[bn,h] = <H[bn,h,:], a_src/dst[h,:]> ----------
__global__ void gat_scores_kernel(const float* __restrict__ Hfeat,
                                  const float* __restrict__ a_src, const float* __restrict__ a_dst,
                                  float* __restrict__ es, float* __restrict__ ed,
                                  int total, int Hh, int Fo) {
    int w = (blockIdx.x * blockDim.x + threadIdx.x) >> 5;
    int lane = threadIdx.x & 31;
    if (w >= total) return;
    int bn = w / Hh;
    int h = w - bn * Hh;
    const float* hp = Hfeat + ((size_t)bn * Hh + h) * Fo;
    const float* as = a_src + h * Fo;
    const float* ad = a_dst + h * Fo;
    float s = 0.0f, d = 0.0f;
    for (int f = lane; f < Fo; f += 32) { float x = hp[f]; s += x * as[f]; d += x * ad[f]; }
    for (int m = 16; m >= 1; m >>= 1) { s += __shfl_xor(s, m, 32); d += __shfl_xor(d, m, 32); }
    if (lane == 0) { es[w] = s; ed[w] = d; }
}

// ---------------- GAT aggregate: segment softmax over incoming edges + weighted sum ------
__global__ void gat_agg_kernel(const float* __restrict__ Hfeat,
                               const float* __restrict__ es, const float* __restrict__ ed,
                               const int* __restrict__ deg, const int* __restrict__ adj,
                               const float* __restrict__ bias,
                               unsigned short* __restrict__ outb,
                               int out_stride, int out_off, int applyElu,
                               int Bn, int Nn, int Hh, int Fo) {
    int w = (blockIdx.x * blockDim.x + threadIdx.x) >> 5;
    int lane = threadIdx.x & 31;
    if (w >= Bn * Nn * Hh) return;
    int h = w % Hh;
    int bn = w / Hh;
    int n = bn % Nn;
    int b = bn / Nn;

    int d = deg[n];
    float edst = ed[((size_t)b * Nn + n) * Hh + h];
    float ev[8];
    int sv[8];
    float m = -3.4e38f;
#pragma unroll
    for (int j = 0; j < 8; ++j) {
        if (j < d) {
            int s = adj[n * 8 + j];
            sv[j] = s;
            float e = es[((size_t)b * Nn + s) * Hh + h] + edst;
            e = (e > 0.0f) ? e : 0.2f * e;   // leaky_relu(0.2)
            ev[j] = e;
            m = fmaxf(m, e);
        }
    }
    float sum = 0.0f;
#pragma unroll
    for (int j = 0; j < 8; ++j) {
        if (j < d) { ev[j] = expf(ev[j] - m); sum += ev[j]; }
    }
    float inv = 1.0f / sum;

    for (int f = lane; f < Fo; f += 32) {
        float acc = 0.0f;
#pragma unroll
        for (int j = 0; j < 8; ++j) {
            if (j < d)
                acc += ev[j] * inv * Hfeat[((size_t)(b * Nn + sv[j]) * Hh + h) * Fo + f];
        }
        float val = acc + bias[h * Fo + f];
        if (applyElu) val = (val > 0.0f) ? val : (expf(val) - 1.0f);
        outb[(size_t)bn * out_stride + out_off + h * Fo + f] = f2b(val);
    }
}

// ---------------- Persistent single-WGP LSTM recurrence --------------------------------
// 1024 threads = 32 wave32 waves. Wave w: unit block u0 = (w>>1)*16, row half m0 = (w&1)*32.
// Per step: g = h_prev @ WhhT (WMMA, bf16 h in LDS, packed B frags) + G[:,t,:].
// Gates are per-lane: WMMA C-layout puts i/f/g/o of one (batch,unit) in the same
// lane/VGPR slot across the 4 gate accumulators. Cell state c lives in registers.
#define HS 264   // 256 + 8 pad ushorts per row
__global__ __launch_bounds__(1024)
void lstm_kernel(const float* __restrict__ G,             // [64*Nn][1024], biases folded in
                 const unsigned short* __restrict__ WhhP, // packed B: K=256, Nc=1024
                 float* __restrict__ Hout,                // [64*Nn][256] f32
                 int Nn) {
    __shared__ __align__(16) unsigned short h_lds[64 * HS];

    const int tid = threadIdx.x;
    const int lane = tid & 31;
    const int wave = tid >> 5;
    const int u0 = (wave >> 1) * 16;
    const int m0 = (wave & 1) * 32;
    const int l15 = lane & 15;
    const int lhi = lane >> 4;
    const int u = u0 + l15;

    for (int i = tid; i < 64 * HS; i += 1024) h_lds[i] = 0;
    float c_reg[2][8];
#pragma unroll
    for (int mi = 0; mi < 2; ++mi)
#pragma unroll
        for (int v = 0; v < 8; ++v) c_reg[mi][v] = 0.0f;
    __syncthreads();

    for (int t = 0; t < Nn; ++t) {
        v8f acc[2][4];
#pragma unroll
        for (int mi = 0; mi < 2; ++mi)
#pragma unroll
            for (int g = 0; g < 4; ++g) acc[mi][g] = {};

#pragma unroll
        for (int kk = 0; kk < 8; ++kk) {
            const int k0 = kk * 32;
            Frag16 af[2];
#pragma unroll
            for (int mi = 0; mi < 2; ++mi) {
                const int row = m0 + mi * 16 + l15;
                const unsigned short* hp = &h_lds[row * HS + k0 + (lhi << 4)];
                af[mi].q[0] = *(const uint4*)(hp);
                af[mi].q[1] = *(const uint4*)(hp + 8);
            }
#pragma unroll
            for (int g = 0; g < 4; ++g) {
                // packed-B fragment: ntiles = 1024/16 = 64
                const unsigned short* bp =
                    WhhP + (((((size_t)kk) * 64 + ((u0 + (g << 8)) >> 4)) * 32 + lane) << 4);
                Frag16 bf;
                bf.q[0] = *(const uint4*)(bp);
                bf.q[1] = *(const uint4*)(bp + 8);
#pragma unroll
                for (int mi = 0; mi < 2; ++mi)
                    acc[mi][g] = __builtin_amdgcn_wmma_f32_16x16x32_bf16(
                        false, af[mi].v, false, bf.v, (short)0, acc[mi][g], false, false);
            }
        }
        __syncthreads();   // everyone done reading h_{t-1}

#pragma unroll
        for (int mi = 0; mi < 2; ++mi) {
#pragma unroll
            for (int v = 0; v < 8; ++v) {
                const int b = m0 + mi * 16 + (lhi << 3) + v;   // batch row
                const size_t gbase = ((size_t)b * Nn + t) * 1024 + u;
                const float pi = acc[mi][0][v] + G[gbase];
                const float pf = acc[mi][1][v] + G[gbase + 256];
                const float pg = acc[mi][2][v] + G[gbase + 512];
                const float po = acc[mi][3][v] + G[gbase + 768];
                const float c = sigf(pf) * c_reg[mi][v] + sigf(pi) * tanhf(pg);
                const float h = sigf(po) * tanhf(c);
                c_reg[mi][v] = c;
                h_lds[b * HS + u] = f2b(h);
                Hout[((size_t)b * Nn + t) * 256 + u] = h;
            }
        }
        __syncthreads();   // h_t visible before next step
    }
}

// ---------------- final projection + sigmoid -------------------------------------------
__global__ void out_kernel(const float* __restrict__ Hout,
                           const int* __restrict__ qn, const int* __restrict__ pn,
                           const float* __restrict__ emb_q, const float* __restrict__ emb_p,
                           const float* __restrict__ Wout, const float* __restrict__ bout,
                           float* __restrict__ y, int BN) {
    int w = (blockIdx.x * blockDim.x + threadIdx.x) >> 5;
    int lane = threadIdx.x & 31;
    if (w >= BN) return;
    const float* hp = Hout + (size_t)w * 256;
    const float* qe = emb_q + (size_t)qn[w] * 256;
    const float* pe = emb_p + (size_t)pn[w] * 256;
    float s = 0.0f;
    for (int f = lane; f < 256; f += 32)
        s += hp[f] * Wout[f] + qe[f] * Wout[256 + f] + pe[f] * Wout[512 + f];
    for (int m = 16; m >= 1; m >>= 1) s += __shfl_xor(s, m, 32);
    if (lane == 0) y[w] = 1.0f / (1.0f + expf(-(s + bout[0])));
}

// ---------------------------------------------------------------------------------------

extern "C" void kernel_launch(void* const* d_in, const int* in_sizes, int n_in,
                              void* d_out, int out_size, void* d_ws, size_t ws_size,
                              hipStream_t stream) {
    const int* p    = (const int*)d_in[1];
    const int* q    = (const int*)d_in[2];
    const int* r    = (const int*)d_in[3];
    const int* aff  = (const int*)d_in[4];
    const int* qn   = (const int*)d_in[5];
    const int* pn   = (const int*)d_in[6];
    const int* esrc = (const int*)d_in[7];
    const int* edst = (const int*)d_in[8];
    const float* emb_p    = (const float*)d_in[9];
    const float* emb_q    = (const float*)d_in[10];
    const float* emb_r    = (const float*)d_in[11];
    const float* emb_aff  = (const float*)d_in[12];
    const float* W_affcat = (const float*)d_in[13];
    const float* b_affcat = (const float*)d_in[14];
    const float* W_g1     = (const float*)d_in[15];
    const float* a_src1   = (const float*)d_in[16];
    const float* a_dst1   = (const float*)d_in[17];
    const float* b_g1     = (const float*)d_in[18];
    const float* W_g2     = (const float*)d_in[19];
    const float* a_src2   = (const float*)d_in[20];
    const float* a_dst2   = (const float*)d_in[21];
    const float* b_g2     = (const float*)d_in[22];
    const float* W_ih     = (const float*)d_in[23];
    const float* W_hh     = (const float*)d_in[24];
    const float* b_ih     = (const float*)d_in[25];
    const float* b_hh     = (const float*)d_in[26];
    const float* W_out    = (const float*)d_in[27];
    const float* b_out    = (const float*)d_in[28];

    const int B  = in_sizes[0];          // 64
    const int BN = in_sizes[1];          // 31936
    const int Nn = BN / B;               // 499
    const int E  = in_sizes[7];          // 1493

    // ---- bump allocator over d_ws ----
    char* cur = (char*)d_ws;
    auto alloc = [&](size_t bytes) -> void* {
        void* ptr = (void*)cur;
        cur += (bytes + 255) & ~(size_t)255;
        return ptr;
    };

    unsigned short* wacP  = (unsigned short*)alloc((size_t)512 * 256 * 2);
    unsigned short* wg1P  = (unsigned short*)alloc((size_t)256 * 1024 * 2);
    unsigned short* wg2P  = (unsigned short*)alloc((size_t)1024 * 256 * 2);
    unsigned short* wihP  = (unsigned short*)alloc((size_t)1024 * 1024 * 2);
    unsigned short* whhP  = (unsigned short*)alloc((size_t)256 * 1024 * 2);
    float*          bsum  = (float*)alloc((size_t)1024 * 4);
    int*            deg   = (int*)alloc((size_t)Nn * 4);
    int*            adj   = (int*)alloc((size_t)Nn * 8 * 4);
    float*          es1   = (float*)alloc((size_t)BN * 8 * 4);
    float*          ed1   = (float*)alloc((size_t)BN * 8 * 4);
    float*          es2   = (float*)alloc((size_t)BN * 4);
    float*          ed2   = (float*)alloc((size_t)BN * 4);
    unsigned short* Xin   = (unsigned short*)alloc((size_t)BN * 512 * 2);
    unsigned short* X0b   = (unsigned short*)alloc((size_t)BN * 256 * 2);
    float*          H2f   = (float*)alloc((size_t)BN * 256 * 4);
    float*          HoutF = (float*)alloc((size_t)BN * 256 * 4);
    // big shared regions (lifetimes disjoint): H1 (GAT1 pre-act) then G (LSTM gates)
    float*          H1f   = (float*)alloc((size_t)BN * 1024 * 4);
    float*          Gf    = H1f;
    // GAT1 output then Seq buffer
    unsigned short* GAT1b = (unsigned short*)alloc((size_t)BN * 1024 * 2);
    unsigned short* SeqB  = GAT1b;

    const int T = 256;
    auto blks = [&](long long n) { return (unsigned)((n + T - 1) / T); };

    // 1) weight packing into WMMA B-fragment layout (+ transpose fold) and bias fold
    pack_b_kernel<<<blks(512 * 256), T, 0, stream>>>(W_affcat, wacP, 512, 256, 0);
    pack_b_kernel<<<blks(256 * 1024), T, 0, stream>>>(W_g1, wg1P, 256, 1024, 0);
    pack_b_kernel<<<blks(1024 * 256), T, 0, stream>>>(W_g2, wg2P, 1024, 256, 0);
    pack_b_kernel<<<blks(1024 * 1024), T, 0, stream>>>(W_ih, wihP, 1024, 1024, 1);
    pack_b_kernel<<<blks(256 * 1024), T, 0, stream>>>(W_hh, whhP, 256, 1024, 1);
    add_bias_kernel<<<blks(1024), T, 0, stream>>>(b_ih, b_hh, bsum, 1024);

    // 2) deterministic CSR of incoming edges
    build_csr_kernel<<<blks(Nn), T, 0, stream>>>(esrc, edst, E, Nn, deg, adj);

    // 3) x = [p_emb | aff_emb] @ W_affcat + b
    build_xin_kernel<<<blks((long long)BN * 512), T, 0, stream>>>(p, aff, emb_p, emb_aff, Xin, BN);
    gemm_bf16_kernel<<<dim3((BN + 63) / 64, 256 / 128), T, 0, stream>>>(
        Xin, wacP, b_affcat, nullptr, X0b, BN, 256, 512);

    // 4) GAT layer 1 (8 heads x 128), ELU
    gemm_bf16_kernel<<<dim3((BN + 63) / 64, 1024 / 128), T, 0, stream>>>(
        X0b, wg1P, nullptr, H1f, nullptr, BN, 1024, 256);
    gat_scores_kernel<<<blks((long long)BN * 8 * 32), T, 0, stream>>>(
        H1f, a_src1, a_dst1, es1, ed1, BN * 8, 8, 128);
    gat_agg_kernel<<<blks((long long)B * Nn * 8 * 32), T, 0, stream>>>(
        H1f, es1, ed1, deg, adj, b_g1, GAT1b, 1024, 0, 1, B, Nn, 8, 128);

    // 5) GAT layer 2 (1 head x 256), writes bf16 straight into Seq[:,768:1024]
    gemm_bf16_kernel<<<dim3((BN + 63) / 64, 256 / 128), T, 0, stream>>>(
        GAT1b, wg2P, nullptr, H2f, nullptr, BN, 256, 1024);
    gat_scores_kernel<<<blks((long long)BN * 32), T, 0, stream>>>(
        H2f, a_src2, a_dst2, es2, ed2, BN, 1, 256);
    gat_agg_kernel<<<blks((long long)B * Nn * 32), T, 0, stream>>>(
        H2f, es2, ed2, deg, adj, b_g2, SeqB, 1024, 768, 0, B, Nn, 1, 256);

    // 6) Seq[:,0:768] = [p_emb | q_emb | r_emb]; hoisted LSTM input projection
    build_seq_kernel<<<blks((long long)BN * 768), T, 0, stream>>>(p, q, r, emb_p, emb_q, emb_r, SeqB, BN);
    gemm_bf16_kernel<<<dim3((BN + 63) / 64, 1024 / 128), T, 0, stream>>>(
        SeqB, wihP, bsum, Gf, nullptr, BN, 1024, 1024);

    // 7) recurrent LSTM on one persistent WGP-resident workgroup
    lstm_kernel<<<1, 1024, 0, stream>>>(Gf, whhP, HoutF, Nn);

    // 8) output head
    out_kernel<<<blks((long long)BN * 32), T, 0, stream>>>(
        HoutF, qn, pn, emb_q, emb_p, W_out, b_out, (float*)d_out, BN);
}
